// SSIMLoss_58514634440991
// MI455X (gfx1250) — matrix operand order repaired and checked by reading, not compiled, not run on hardware
//
#include <hip/hip_runtime.h>

typedef __attribute__((ext_vector_type(16))) _Float16 v16h;
typedef __attribute__((ext_vector_type(2)))  _Float16 h2t;
typedef __attribute__((ext_vector_type(8)))  float    v8f;
typedef __attribute__((ext_vector_type(8)))  int      v8i;

#define IMG_H   512
#define IMG_W   512
#define TILES_X 32
#define TILES_Y 32
#define NPLANES 96                              // 32 images * 3 channels
#define NTILES  (NPLANES * TILES_Y * TILES_X)   // 98304
#define SSIM_C1 0.0001f
#define SSIM_C2 0.0009f

// Normalized 1D Gaussian, sigma=1.5, 11 taps (g[d] = exp(-(d-5)^2/4.5)/S).
constexpr double G11[11] = {
    0.00102838, 0.00759875, 0.03600077, 0.10936069, 0.21300553,
    0.26601172, 0.21300553, 0.10936069, 0.03600077, 0.00759875, 0.00102838
};

// Per-lane WMMA fragments of the two constant band matrices, built at
// compile time.
//  Bh (32x16 B layout): K = e + 8*(e>=8) + 8*hi, N = n16,  Bh[k][c] = g[k - c - 3]
//  Av (16x32 A layout): M = n16, K = e + 16*hi, Av[m][k] = g[k - m]
struct alignas(32) FragTabs {
    _Float16 bh[32][16];
    _Float16 av[32][16];
};

constexpr FragTabs make_tabs() {
    FragTabs t{};
    for (int lane = 0; lane < 32; ++lane) {
        const int hi = lane >> 4;
        const int n  = lane & 15;
        for (int e = 0; e < 16; ++e) {
            const int kb = e + ((e >> 3) << 3) + 8 * hi;
            const int db = kb - n - 3;
            t.bh[lane][e] = (db >= 0 && db <= 10) ? (_Float16)G11[db]
                                                  : (_Float16)0.0;
            const int ka = e + 16 * hi;
            const int da = ka - n;
            t.av[lane][e] = (da >= 0 && da <= 10) ? (_Float16)G11[da]
                                                  : (_Float16)0.0;
        }
    }
    return t;
}

__constant__ FragTabs FT = make_tabs();

__device__ __forceinline__ int clampi(int v, int lo, int hi) {
    return v < lo ? lo : (v > hi ? hi : v);
}

// ds_swizzle group-of-32 xor pattern: src lane = lane ^ mask
template <int IMM>
__device__ __forceinline__ float swz(float x) {
    return __int_as_float(__builtin_amdgcn_ds_swizzle(__float_as_int(x), IMM));
}
#define SWZ_XOR(mask) (((mask) << 10) | 0x1f)

__device__ __forceinline__ void pack8(v16h& dst, int base, float4 a, float4 b) {
    dst[base + 0] = (_Float16)a.x; dst[base + 1] = (_Float16)a.y;
    dst[base + 2] = (_Float16)a.z; dst[base + 3] = (_Float16)a.w;
    dst[base + 4] = (_Float16)b.x; dst[base + 5] = (_Float16)b.y;
    dst[base + 6] = (_Float16)b.z; dst[base + 7] = (_Float16)b.w;
}

// One wave per 16x16 output tile. Fused: load haloed tile -> 5 channels
// -> horizontal blur (WMMA) -> swizzle repack -> vertical blur (WMMA)
// -> SSIM -> wave-reduced partial sum.
__global__ __launch_bounds__(32)
void ssim_tile_kernel(const float* __restrict__ img1,
                      const float* __restrict__ img2,
                      float* __restrict__ partial) {
    const int tile = blockIdx.x;               // 0..NTILES-1
    const int txy  = tile & (TILES_X * TILES_Y - 1);
    const int pl   = tile >> 10;               // plane index (img*3+ch)
    const int tx   = txy & (TILES_X - 1);
    const int ty   = txy >> 5;

    const size_t plane_off = (size_t)pl * (IMG_H * IMG_W);
    const int lane = threadIdx.x;
    const int hi   = lane >> 4;                // 0: lanes 0-15, 1: lanes 16-31
    const int n16  = lane & 15;

    const int row0    = ty * 16 - 5;           // row halo = 5 (rows 0..25 used)
    const int colbase = tx * 16 - 8;           // col halo = 8 (32B aligned)

    // ---- Load per-lane rows of both images into f16 A-fragment layout ----
    // A layout: M = lane&15 (+16 per chunk), K(e) = e + 8*(e>=8) + 8*hi
    v16h h1[2], h2[2];
    const bool interior = (tx >= 1) && (tx <= TILES_X - 2) &&
                          (ty >= 1) && (ty <= TILES_Y - 2);
    if (interior) {
        // All touched rows/cols are in-image; out-of-band rows (26-31) and
        // cols (k<=2, k>=29) get exactly-zero band weights, so no masking.
#pragma unroll
        for (int ck = 0; ck < 2; ++ck) {
            const int rg = row0 + n16 + 16 * ck;
            const float* rp1 = img1 + plane_off + (size_t)rg * IMG_W + colbase;
            const float* rp2 = img2 + plane_off + (size_t)rg * IMG_W + colbase;
            const float4* q1a = (const float4*)(rp1 + (hi << 3));
            const float4* q1b = (const float4*)(rp1 + 16 + (hi << 3));
            const float4* q2a = (const float4*)(rp2 + (hi << 3));
            const float4* q2b = (const float4*)(rp2 + 16 + (hi << 3));
            pack8(h1[ck], 0, q1a[0], q1a[1]);
            pack8(h1[ck], 8, q1b[0], q1b[1]);
            pack8(h2[ck], 0, q2a[0], q2a[1]);
            pack8(h2[ck], 8, q2b[0], q2b[1]);
        }
    } else {
        // Border tiles: zero-padding outside the image (SAME padding).
#pragma unroll
        for (int ck = 0; ck < 2; ++ck) {
            const int rg   = row0 + n16 + 16 * ck;
            const bool rok = (rg >= 0) && (rg < IMG_H);
            const int rcl  = clampi(rg, 0, IMG_H - 1);
            const float* r1 = img1 + plane_off + (size_t)rcl * IMG_W;
            const float* r2 = img2 + plane_off + (size_t)rcl * IMG_W;
#pragma unroll
            for (int e = 0; e < 16; ++e) {
                const int k   = e + ((e >> 3) << 3) + (hi << 3);
                const int cg  = colbase + k;
                const bool ok = rok && (cg >= 0) && (cg < IMG_W);
                const int ccl = clampi(cg, 0, IMG_W - 1);
                const float u = r1[ccl];
                const float v = r2[ccl];
                h1[ck][e] = (_Float16)(ok ? u : 0.0f);
                h2[ck][e] = (_Float16)(ok ? v : 0.0f);
            }
        }
    }

    // ---- Constant band fragments from compile-time table ----
    const v16h Bh = *(const v16h*)&FT.bh[lane][0];
    const v16h Av = *(const v16h*)&FT.av[lane][0];

    const v8f zero_c = {};
    v8f blur[5];

    // ---- Per channel: horizontal WMMA x2, swizzle repack, vertical WMMA ----
#pragma unroll
    for (int p = 0; p < 5; ++p) {
        v16h s0, s1;
        if      (p == 0) { s0 = h1[0];         s1 = h1[1];         }
        else if (p == 1) { s0 = h2[0];         s1 = h2[1];         }
        else if (p == 2) { s0 = h1[0] * h1[0]; s1 = h1[1] * h1[1]; }
        else if (p == 3) { s0 = h2[0] * h2[0]; s1 = h2[1] * h2[1]; }
        else             { s0 = h1[0] * h2[0]; s1 = h1[1] * h2[1]; }

        // Horizontal blur: row chunks 0-15 and 16-31
        v8f d0 = __builtin_amdgcn_wmma_f32_16x16x32_f16(
            false, s0, false, Bh, (short)0, zero_c, false, false);
        v8f d1 = __builtin_amdgcn_wmma_f32_16x16x32_f16(
            false, s1, false, Bh, (short)0, zero_c, false, false);

        // Repack C/D layout (row = r + 8*hi [+16 chunk1], col = n16) into
        // 32x16 B layout (row = e + 16*hi, col = n16). Pack row-adjacent f16
        // pairs, swap across lane^16 with ds_swizzle, then each b-fragment
        // dword is one 32-bit select:
        //   hi=0: dword t   = own_lo (rows 2t,2t+1), dword 4+t = got_lo (8+2t)
        //   hi=1: dword t   = got_hi (rows 16+2t),   dword 4+t = own_hi (24+2t)
        v8i bi;
#pragma unroll
        for (int t = 0; t < 4; ++t) {
            h2t plo, phi;
            plo.x = (_Float16)d0[2 * t];     plo.y = (_Float16)d0[2 * t + 1];
            phi.x = (_Float16)d1[2 * t];     phi.y = (_Float16)d1[2 * t + 1];
            const int olo = __builtin_bit_cast(int, plo);
            const int ohi = __builtin_bit_cast(int, phi);
            const int glo = __builtin_amdgcn_ds_swizzle(olo, SWZ_XOR(0x10));
            const int ghi = __builtin_amdgcn_ds_swizzle(ohi, SWZ_XOR(0x10));
            bi[t]     = hi ? ghi : olo;
            bi[4 + t] = hi ? ohi : glo;
        }
        const v16h b = __builtin_bit_cast(v16h, bi);

        // Vertical blur: out = Av(16x32 band) x H(32x16)
        blur[p] = __builtin_amdgcn_wmma_f32_16x16x32_f16(
            false, Av, false, b, (short)0, zero_c, false, false);
    }

    // ---- SSIM map over this lane's 8 output pixels ----
    float s = 0.0f;
#pragma unroll
    for (int i = 0; i < 8; ++i) {
        const float m1  = blur[0][i];
        const float m2  = blur[1][i];
        const float m1s = m1 * m1;
        const float m2s = m2 * m2;
        const float m12 = m1 * m2;
        const float s1  = blur[2][i] - m1s;
        const float s2  = blur[3][i] - m2s;
        const float s12 = blur[4][i] - m12;
        const float num = (2.0f * m12 + SSIM_C1) * (2.0f * s12 + SSIM_C2);
        const float den = (m1s + m2s + SSIM_C1) * (s1 + s2 + SSIM_C2);
        s += num * __builtin_amdgcn_rcpf(den);
    }

    // ---- Wave reduction via ds_swizzle xor tree (deterministic) ----
    s += swz<SWZ_XOR(0x10)>(s);
    s += swz<SWZ_XOR(0x08)>(s);
    s += swz<SWZ_XOR(0x04)>(s);
    s += swz<SWZ_XOR(0x02)>(s);
    s += swz<SWZ_XOR(0x01)>(s);
    if (lane == 0)
        partial[tile] = s;
}

// Deterministic final reduction: fixed per-thread strided order + fixed tree,
// f64 accumulation (sum ~ 25.2M; f32 would lose low bits).
__global__ __launch_bounds__(256)
void ssim_reduce_kernel(const float* __restrict__ partial,
                        int n,
                        float* __restrict__ out) {
    __shared__ double sm[256];
    double acc = 0.0;
    for (int i = threadIdx.x; i < n; i += 256)
        acc += (double)partial[i];
    sm[threadIdx.x] = acc;
    __syncthreads();
#pragma unroll
    for (int stride = 128; stride > 0; stride >>= 1) {
        if (threadIdx.x < (unsigned)stride)
            sm[threadIdx.x] += sm[threadIdx.x + stride];
        __syncthreads();
    }
    if (threadIdx.x == 0) {
        const double inv_n = 1.0 / (double)((size_t)NPLANES * IMG_H * IMG_W);
        out[0] = (float)(1.0 - sm[0] * inv_n);
    }
}

extern "C" void kernel_launch(void* const* d_in, const int* in_sizes, int n_in,
                              void* d_out, int out_size, void* d_ws, size_t ws_size,
                              hipStream_t stream) {
    (void)in_sizes; (void)n_in; (void)out_size; (void)ws_size;
    const float* img1 = (const float*)d_in[0];
    const float* img2 = (const float*)d_in[1];
    float* partial = (float*)d_ws;              // NTILES * 4 bytes = 384 KB
    float* out = (float*)d_out;

    ssim_tile_kernel<<<NTILES, 32, 0, stream>>>(img1, img2, partial);
    ssim_reduce_kernel<<<1, 256, 0, stream>>>(partial, NTILES, out);
}
